// GraphSelfAttention_39900246180618
// MI455X (gfx1250) — matrix operand order repaired and checked by reading, not compiled, not run on hardware
//
#include <hip/hip_runtime.h>
#include <hip/hip_bf16.h>
#include <stdint.h>

// ---------------------------------------------------------------------------
// Graph self-attention, N=8192, D=512, fp32 reference (MI455X / gfx1250).
// Pipeline (all bf16 WMMA with f32 accumulation):
//   1) cast X, Wq, Wk, Wv, Wo -> bf16
//   2) Qb = (X Wq^T) * 1/sqrt(D)           [bf16]
//   3) Kb =  X Wk^T                        [bf16]
//   4) VbT = (X Wv^T)^T                    [bf16, transposed store]
//   5) S  = mask(Qb Kb^T, adj)             [bf16, -inf where adj==0]
//   6) softmax rows of S in place          [f32 math]
//   7) O1 = S @ V   (A=S, BT=VbT)          [bf16]
//   8) out = O1 Wo^T + bo                  [f32 -> d_out]
// GEMM: 128(M)x256(N) block tile, 8 wave32s (2Mx4N), 64x64 wave tile
// (4x4 WMMA). K staged in 32-chunks through TRIPLE-buffered LDS with
// GLOBAL_LOAD_ASYNC_TO_LDS_B128 (ASYNCcnt); one workgroup barrier per
// K-step; fragments read via ds_load_b128 (16 loads : 16 WMMAs).
// ---------------------------------------------------------------------------

typedef __attribute__((ext_vector_type(8)))  float  v8f;
typedef __attribute__((ext_vector_type(8)))  __bf16 v8bf;
typedef __attribute__((ext_vector_type(16))) __bf16 v16bf;

#define BROW 40   // LDS row stride in bf16 for a 32-wide K slice (32 + 8 pad)

__device__ __forceinline__ v16bf cat16(v8bf lo, v8bf hi) {
  return __builtin_shufflevector(lo, hi, 0, 1, 2, 3, 4, 5, 6, 7,
                                         8, 9, 10, 11, 12, 13, 14, 15);
}

__device__ __forceinline__ v8f wmma_bf16(v16bf a, v16bf b, v8f c) {
  return __builtin_amdgcn_wmma_f32_16x16x32_bf16(
      false, a, false, b, (short)0, c, false, false);
}

// Async global->LDS 16B copy, per-lane addressing, tracked by ASYNCcnt.
__device__ __forceinline__ void async_copy_b128(unsigned ldsOff,
                                                const __bf16* g) {
  asm volatile("global_load_async_to_lds_b128 %0, %1, off"
               :: "v"(ldsOff), "v"(g)
               : "memory");
}

template <int N>
__device__ __forceinline__ void wait_asynccnt() {
  asm volatile("s_wait_asynccnt %0" :: "i"(N) : "memory");
}

// A fragment (16x32 bf16) from LDS tile: lane L -> row L%16, half h = L/16,
// K-runs h*8+[0..7] and 16+h*8+[0..7]  (ISA 7.12.2 16-bit A layout).
__device__ __forceinline__ v16bf lds_frag_a(const __bf16* s, int row, int h) {
  const v8bf* p = (const v8bf*)(s + row * BROW + h * 8);
  return cat16(p[0], p[2]);
}

// B fragment (32x16 bf16) from LDS tile of B^T: lane L -> col L%16,
// K-runs h*8+[0..7] and 16+h*8+[0..7] (same order as A).
__device__ __forceinline__ v16bf lds_frag_b(const __bf16* s, int row, int h) {
  const v8bf* p = (const v8bf*)(s + row * BROW + h * 8);
  return cat16(p[0], p[2]);
}

// ---- GEMM: C[M,Nc] = scale * (A @ BT^T) (+bias) (+mask) -------------------
// A : bf16 [M, K] row-major.  BT: bf16 [Nc, K] row-major (B transposed).
template <bool OUT_BF16, bool TRANS_OUT, bool MASK, bool BIAS>
__global__ __launch_bounds__(256)
void gemm_bf16_kernel(const __bf16* __restrict__ A,
                      const __bf16* __restrict__ BT,
                      void* __restrict__ Cout,
                      const int* __restrict__ adj,
                      const float* __restrict__ bias,
                      int M, int Nc, int K, float scale) {
  __shared__ __bf16 As[3][128 * BROW];   // 3 x 10240 B
  __shared__ __bf16 Bs[3][256 * BROW];   // 3 x 20480 B

  const int t    = threadIdx.x;
  const int lane = t & 31;
  const int w    = t >> 5;    // 0..7
  const int wm   = w & 1;     // 0..1 -> M offset wm*64
  const int wn   = w >> 1;    // 0..3 -> N offset wn*64
  const int mBase = blockIdx.y * 128;
  const int nBase = blockIdx.x * 256;

  const __bf16* __restrict__ Ablk = A  + (size_t)mBase * K;
  const __bf16* __restrict__ Bblk = BT + (size_t)nBase * K;

  // Staging: A tile 128x32 bf16 = 512 x 16B chunks (2/thread),
  //          B tile 256x32 bf16 = 1024 x 16B chunks (4/thread)
  // => 6 async ops per thread per stage.
  const int cq = t & 3;        // 16B chunk within the 64B K-row slice
  const int r0 = t >> 2;       // 0..63

  const __bf16* gA[2];
  const __bf16* gB[4];
  unsigned lA[2], lB[4];
#pragma unroll
  for (int i = 0; i < 2; ++i) {
    gA[i] = Ablk + (size_t)(r0 + 64 * i) * K + cq * 8;
    lA[i] = (unsigned)(uintptr_t)&As[0][(r0 + 64 * i) * BROW + cq * 8];
  }
#pragma unroll
  for (int i = 0; i < 4; ++i) {
    gB[i] = Bblk + (size_t)(r0 + 64 * i) * K + cq * 8;
    lB[i] = (unsigned)(uintptr_t)&Bs[0][(r0 + 64 * i) * BROW + cq * 8];
  }
  const unsigned aBufB = 128 * BROW * 2;   // bytes per A buffer
  const unsigned bBufB = 256 * BROW * 2;   // bytes per B buffer

  auto stage = [&](int ks, int buf) {
    const size_t go = (size_t)ks * 32;
    const unsigned ao = (unsigned)buf * aBufB;
    const unsigned bo = (unsigned)buf * bBufB;
#pragma unroll
    for (int i = 0; i < 2; ++i) async_copy_b128(lA[i] + ao, gA[i] + go);
#pragma unroll
    for (int i = 0; i < 4; ++i) async_copy_b128(lB[i] + bo, gB[i] + go);
  };

  v8f acc[4][4] = {};
  const int h  = lane >> 4;
  const int nl = lane & 15;
  const int nk = K / 32;

  stage(0, 0);
  stage(1, 1);
  for (int k = 0; k < nk; ++k) {
    if (k + 1 < nk) wait_asynccnt<6>();   // stage k landed (in-order)
    else            wait_asynccnt<0>();
    __syncthreads();                      // all waves see buffer k%3
    if (k + 2 < nk) stage(k + 2, (k + 2) % 3);  // buf (k+2)%3 == (k-1)%3:
                                                // last read finished pre-barrier
    const __bf16* Ac = As[k % 3];
    const __bf16* Bc = Bs[k % 3];
    v16bf af[4], bfr[4];
#pragma unroll
    for (int tm = 0; tm < 4; ++tm)
      af[tm] = lds_frag_a(Ac, wm * 64 + tm * 16 + nl, h);
#pragma unroll
    for (int tn = 0; tn < 4; ++tn)
      bfr[tn] = lds_frag_b(Bc, wn * 64 + tn * 16 + nl, h);
#pragma unroll
    for (int tm = 0; tm < 4; ++tm)
#pragma unroll
      for (int tn = 0; tn < 4; ++tn)
        acc[tm][tn] = wmma_bf16(af[tm], bfr[tn], acc[tm][tn]);
  }

  // Epilogue. C/D layout: lane L -> column n = L%16; VGPR r -> row r + 8*(L/16).
#pragma unroll
  for (int tm = 0; tm < 4; ++tm) {
#pragma unroll
    for (int tn = 0; tn < 4; ++tn) {
      const int gn = nBase + wn * 64 + tn * 16 + nl;
      float bval = 0.0f;
      if (BIAS) bval = bias[gn];
#pragma unroll
      for (int r = 0; r < 8; ++r) {
        const int gm = mBase + wm * 64 + tm * 16 + h * 8 + r;
        float val = acc[tm][tn][r] * scale + bval;
        if (MASK) {
          if (adj[(size_t)gm * Nc + gn] == 0) val = -__builtin_inff();
        }
        if (TRANS_OUT) {
          ((__bf16*)Cout)[(size_t)gn * M + gm] = (__bf16)val;
        } else if (OUT_BF16) {
          ((__bf16*)Cout)[(size_t)gm * Nc + gn] = (__bf16)val;
        } else {
          ((float*)Cout)[(size_t)gm * Nc + gn] = val;
        }
      }
    }
  }
}

// ---- fp32 -> bf16 cast ----------------------------------------------------
__global__ __launch_bounds__(256)
void f32_to_bf16_kernel(const float* __restrict__ src,
                        __bf16* __restrict__ dst, int n) {
  int i = blockIdx.x * blockDim.x + threadIdx.x;
  if (i < n) dst[i] = (__bf16)src[i];
}

// ---- in-place row softmax on bf16 scores (f32 math) -----------------------
__global__ __launch_bounds__(256)
void softmax_rows_kernel(__bf16* __restrict__ S, int N) {
  const int row = blockIdx.x;
  const int t   = threadIdx.x;
  __bf16* Srow = S + (size_t)row * N;

  float v[32];
  float m = -__builtin_inff();
#pragma unroll
  for (int i = 0; i < 32; ++i) {
    v[i] = (float)Srow[t + i * 256];
    m = fmaxf(m, v[i]);
  }

  __shared__ float red[256];
  red[t] = m;
  __syncthreads();
  for (int s = 128; s > 0; s >>= 1) {
    if (t < s) red[t] = fmaxf(red[t], red[t + s]);
    __syncthreads();
  }
  m = red[0];
  __syncthreads();

  float l = 0.0f;
#pragma unroll
  for (int i = 0; i < 32; ++i) {
    v[i] = __expf(v[i] - m);   // exp(-inf - m) == 0 for masked entries
    l += v[i];
  }
  red[t] = l;
  __syncthreads();
  for (int s = 128; s > 0; s >>= 1) {
    if (t < s) red[t] += red[t + s];
    __syncthreads();
  }
  l = red[0];

  const float inv = 1.0f / l;
#pragma unroll
  for (int i = 0; i < 32; ++i) {
    Srow[t + i * 256] = (__bf16)(v[i] * inv);
  }
}

// ---------------------------------------------------------------------------
extern "C" void kernel_launch(void* const* d_in, const int* in_sizes, int n_in,
                              void* d_out, int out_size, void* d_ws, size_t ws_size,
                              hipStream_t stream) {
  (void)in_sizes; (void)n_in; (void)out_size; (void)ws_size;

  const int N = 8192;
  const int D = 512;

  const float* X   = (const float*)d_in[0];
  const int*   adj = (const int*)d_in[1];
  const float* Wq  = (const float*)d_in[2];
  const float* Wk  = (const float*)d_in[3];
  const float* Wv  = (const float*)d_in[4];
  const float* Wo  = (const float*)d_in[5];
  const float* bo  = (const float*)d_in[6];
  float* out = (float*)d_out;

  // workspace carve (256B aligned)
  char* ws = (char*)d_ws;
  auto carve = [&](size_t bytes) -> char* {
    char* p = ws;
    ws += (bytes + 255) & ~(size_t)255;
    return p;
  };
  __bf16* Xb  = (__bf16*)carve((size_t)N * D * 2);
  __bf16* Wqb = (__bf16*)carve((size_t)D * D * 2);
  __bf16* Wkb = (__bf16*)carve((size_t)D * D * 2);
  __bf16* Wvb = (__bf16*)carve((size_t)D * D * 2);
  __bf16* Wob = (__bf16*)carve((size_t)D * D * 2);
  __bf16* Qb  = (__bf16*)carve((size_t)N * D * 2);
  __bf16* Kb  = (__bf16*)carve((size_t)N * D * 2);
  __bf16* VbT = (__bf16*)carve((size_t)N * D * 2);   // [D, N] = V^T
  __bf16* O1b = (__bf16*)carve((size_t)N * D * 2);
  __bf16* S   = (__bf16*)carve((size_t)N * N * 2);   // scores -> attn (in place)

  // 1) casts
  f32_to_bf16_kernel<<<(N * D) / 256, 256, 0, stream>>>(X, Xb, N * D);
  f32_to_bf16_kernel<<<(D * D) / 256, 256, 0, stream>>>(Wq, Wqb, D * D);
  f32_to_bf16_kernel<<<(D * D) / 256, 256, 0, stream>>>(Wk, Wkb, D * D);
  f32_to_bf16_kernel<<<(D * D) / 256, 256, 0, stream>>>(Wv, Wvb, D * D);
  f32_to_bf16_kernel<<<(D * D) / 256, 256, 0, stream>>>(Wo, Wob, D * D);

  const dim3 blk(256);
  const dim3 gridProj(D / 256, N / 128);   // (2, 64)
  const dim3 gridScore(N / 256, N / 128);  // (32, 64)
  const float qscale = 0.04419417382415922f;  // 1/sqrt(512)

  // 2) Qb = (X Wq^T) * qscale   (row-major weights == B^T layout)
  gemm_bf16_kernel<true, false, false, false><<<gridProj, blk, 0, stream>>>(
      Xb, Wqb, Qb, nullptr, nullptr, N, D, D, qscale);
  // 3) Kb = X Wk^T
  gemm_bf16_kernel<true, false, false, false><<<gridProj, blk, 0, stream>>>(
      Xb, Wkb, Kb, nullptr, nullptr, N, D, D, 1.0f);
  // 4) VbT = (X Wv^T)^T
  gemm_bf16_kernel<true, true, false, false><<<gridProj, blk, 0, stream>>>(
      Xb, Wvb, VbT, nullptr, nullptr, N, D, D, 1.0f);
  // 5) S = mask(Qb Kb^T)
  gemm_bf16_kernel<true, false, true, false><<<gridScore, blk, 0, stream>>>(
      Qb, Kb, S, adj, nullptr, N, N, D, 1.0f);
  // 6) softmax rows in place
  softmax_rows_kernel<<<N, 256, 0, stream>>>(S, N);
  // 7) O1 = attn @ V   (A = S [N,N], BT = V^T [D,N])
  gemm_bf16_kernel<true, false, false, false><<<gridProj, blk, 0, stream>>>(
      S, VbT, O1b, nullptr, nullptr, N, D, N, 1.0f);
  // 8) out = O1 Wo^T + bo  -> f32 d_out
  gemm_bf16_kernel<false, false, false, true><<<gridProj, blk, 0, stream>>>(
      O1b, Wob, out, nullptr, bo, N, D, D, 1.0f);
}
